// ImplicitFlowDecoder_45286135169671
// MI455X (gfx1250) — matrix operand order repaired and checked by reading, not compiled, not run on hardware
//
#include <hip/hip_runtime.h>
#include <hip/hip_bf16.h>

typedef _Float16 v16h __attribute__((ext_vector_type(16)));
typedef _Float16 v8h  __attribute__((ext_vector_type(8)));
typedef float    v8f  __attribute__((ext_vector_type(8)));

#define B_    2
#define H8_   48
#define W8_   64
#define H16_  24
#define W16_  32
#define HF_   384
#define WF_   512
#define NPIX  (B_*HF_*WF_)
#define TILE  64
#define NTHR  128
#define PPF   24          // floats of per-pixel params

// LDS layout (units: halves)
#define ACTA_OFF   0
#define ACTB_OFF   (ACTA_OFF + TILE*256)
#define SKIP_OFF   (ACTB_OFF + TILE*256)
#define F1W_OFF    (SKIP_OFF + TILE*128)
#define EXT_OFF    (F1W_OFF  + TILE*128)
#define WT_OFF     (EXT_OFF  + TILE*32)
#define HALVES_TOT (WT_OFF   + 256*288)
#define LDS_BYTES  (HALVES_TOT*2 + TILE*PPF*4)

#define VZERO {0.f, 0.f, 0.f, 0.f, 0.f, 0.f, 0.f, 0.f}

struct Params {
  const float *feat_s8, *feat1_s8, *feat_s16, *ctx_s8, *coarse_flow;
  const float *proj_ctx_w, *proj_ctx_b, *gate1;
  const float *ffn1_w1, *ffn1_b1, *ffn1_w2, *ffn1_b2;
  const float *proj_s8_w, *proj_s8_b, *gate2;
  const float *ffn2_w1, *ffn2_b1, *ffn2_w2, *ffn2_b2;
  const float *hw0, *hb0, *hw1, *hb1, *hw2, *hb2, *hw3, *hb3;
  float* out;
};

__device__ __forceinline__ float clipc(float v) {
  return fminf(fmaxf(v, -1.f + 1e-6f), 1.f - 1e-6f);
}
__device__ __forceinline__ float gelu_f(float x) {
  return 0.5f * x * (1.f + erff(x * 0.70710678118654752f));
}
// fast sigmoid: v_exp + v_rcp instead of the IEEE divide sequence
__device__ __forceinline__ float sigmoid_f(float x) {
  return __builtin_amdgcn_rcpf(1.f + __expf(-x));
}

// L2 prefetch of a weight array (emits global_prefetch_b8), cooperates across block
__device__ __forceinline__ void prefetch_bytes(const void* p, int bytes) {
  const char* c = (const char*)p;
  for (int off = (int)threadIdx.x * 128; off < bytes; off += NTHR * 128)
    __builtin_prefetch(c + off, 0, 3);
}

// bilinear grid params for one sampling grid
__device__ __forceinline__ void grid_params(float cx, float cy, int Wf, int Hf, float* q) {
  float x = (cx + 1.f) * (float)Wf * 0.5f - 0.5f;
  float y = (cy + 1.f) * (float)Hf * 0.5f - 0.5f;
  float xf = floorf(x), yf = floorf(y);
  int x0 = min(max((int)xf, 0), Wf - 1);
  int x1 = min(max((int)xf + 1, 0), Wf - 1);
  int y0 = min(max((int)yf, 0), Hf - 1);
  int y1 = min(max((int)yf + 1, 0), Hf - 1);
  q[0] = __int_as_float(x0); q[1] = __int_as_float(x1);
  q[2] = __int_as_float(y0); q[3] = __int_as_float(y1);
  q[4] = x - xf; q[5] = y - yf;
}

// cooperative bilinear gather of C channels for 64 pixels -> f16 LDS buffer
__device__ __forceinline__ void sample_feat(_Float16* dst, int dstStride, const float* fm,
                                            int C, int Hf, int Wf, const float* pp, int go) {
  for (int j = threadIdx.x; j < TILE * C; j += NTHR) {
    int m = j & (TILE - 1);
    int ch = j >> 6;
    const float* q = pp + m * PPF;
    int bi = __float_as_int(q[20]);
    int x0 = __float_as_int(q[go + 0]), x1 = __float_as_int(q[go + 1]);
    int y0 = __float_as_int(q[go + 2]), y1 = __float_as_int(q[go + 3]);
    float wx = q[go + 4], wy = q[go + 5];
    const float* base = fm + ((size_t)(bi * C + ch)) * Hf * Wf;
    float v00 = base[y0 * Wf + x0], v01 = base[y0 * Wf + x1];
    float v10 = base[y1 * Wf + x0], v11 = base[y1 * Wf + x1];
    float val = v00 * (1.f - wx) * (1.f - wy) + v01 * wx * (1.f - wy)
              + v10 * (1.f - wx) * wy + v11 * wx * wy;
    dst[m * dstStride + ch] = (_Float16)val;
  }
}

// stage weight matrix W[K][Nreal] (row-major, (in,out)) transposed into LDS as f16 [Npad][kPad]
__device__ __forceinline__ void stage_wT(_Float16* wT, const float* W,
                                         int K, int Nreal, int Npad, int kPad) {
  for (int j = threadIdx.x; j < Npad * kPad; j += NTHR) {
    int n = j / kPad;
    int k = j - n * kPad;
    float v = (k < K && n < Nreal) ? W[k * Nreal + n] : 0.f;
    wT[j] = (_Float16)v;
  }
}

// A fragment: 16x32 f16. lane m = lane&15; g = lane>>4; halves: K = kb+8g.. , kb+16+8g..
template <int SSTR>
__device__ __forceinline__ v16h load_a(const _Float16* src, int mBase, int kb) {
  int lane = threadIdx.x & 31;
  int row = mBase + (lane & 15);
  int g = lane >> 4;
  const _Float16* p = src + row * SSTR + kb + 8 * g;
  v8h lo = *(const v8h*)p;
  v8h hi = *(const v8h*)(p + 16);
  v16h r;
#pragma unroll
  for (int i = 0; i < 8; ++i) { r[i] = lo[i]; r[8 + i] = hi[i]; }
  return r;
}

// B fragment: 32x16 f16 from transposed weights wT[N][kPad]. lane n = lane&15; halves: K = k+8g.. , k+16+8g..
template <int KPAD>
__device__ __forceinline__ v16h load_b(const _Float16* wT, int nBase, int k) {
  int lane = threadIdx.x & 31;
  int n = nBase + (lane & 15);
  int g = lane >> 4;
  const _Float16* p = wT + n * KPAD + k + 8 * g;
  v8h lo = *(const v8h*)p;
  v8h hi = *(const v8h*)(p + 16);
  v16h r;
#pragma unroll
  for (int i = 0; i < 8; ++i) { r[i] = lo[i]; r[8 + i] = hi[i]; }
  return r;
}

// single-output-tile accumulate (used for the final N=16 layer)
template <int K, int SSTR, int KPAD>
__device__ __forceinline__ void gemm_acc(v8f& acc, const _Float16* src, int mBase,
                                         const _Float16* wT, int kOff, int nBase) {
#pragma unroll
  for (int kb = 0; kb < K; kb += 32) {
    v16h a = load_a<SSTR>(src, mBase, kb);
    v16h b = load_b<KPAD>(wT, nBase, kOff + kb);
    acc = __builtin_amdgcn_wmma_f32_16x16x32_f16(false, a, false, b, (short)0, acc,
                                                 false, false);
  }
}

// dual-output-tile accumulate: shares the A fragment across two N tiles
// (halves A-side LDS traffic, two independent WMMA chains for XDL pipelining)
template <int K, int SSTR, int KPAD>
__device__ __forceinline__ void gemm2_acc(v8f& acc0, v8f& acc1, const _Float16* src,
                                          int mBase, const _Float16* wT, int kOff,
                                          int nBase) {
#pragma unroll
  for (int kb = 0; kb < K; kb += 32) {
    v16h a  = load_a<SSTR>(src, mBase, kb);
    v16h b0 = load_b<KPAD>(wT, nBase,      kOff + kb);
    v16h b1 = load_b<KPAD>(wT, nBase + 16, kOff + kb);
    acc0 = __builtin_amdgcn_wmma_f32_16x16x32_f16(false, a, false, b0, (short)0, acc0,
                                                  false, false);
    acc1 = __builtin_amdgcn_wmma_f32_16x16x32_f16(false, a, false, b1, (short)0, acc1,
                                                  false, false);
  }
}

__global__ void __launch_bounds__(NTHR, 1)
ifd_kernel(Params P) {
  extern __shared__ char smem[];
  _Float16* lds  = (_Float16*)smem;
  _Float16* actA = lds + ACTA_OFF;   // fctx(64) -> t1(256) -> h3(128) -> fused(128) -> h(128)
  _Float16* actB = lds + ACTB_OFF;   // f8/h1(128) -> h2(128) -> t2(256) -> h(256) -> h(64)
  _Float16* skp  = lds + SKIP_OFF;   // f16 samples (128)
  _Float16* f1w  = lds + F1W_OFF;    // warped feat1 samples (128)
  _Float16* ext  = lds + EXT_OFF;    // [cx, cy, cnx, cny, 0...] (32, zero padded)
  _Float16* wT   = lds + WT_OFF;     // transposed weights, up to 256x288
  float*    pp   = (float*)(lds + HALVES_TOT);

  const int tid = threadIdx.x;
  const int tileBase = blockIdx.x * TILE;

  // ---------- phase 0: per-pixel sampling params + coarse flow ----------
  if (tid < TILE) {
    int p = tileBase + tid;
    int bi = p / (HF_ * WF_);
    int rem = p - bi * (HF_ * WF_);
    int h = rem >> 9;
    int w = rem & (WF_ - 1);
    float cx = clipc(2.f * ((float)w + 0.5f) / (float)WF_ - 1.f);
    float cy = clipc(2.f * ((float)h + 0.5f) / (float)HF_ - 1.f);
    float* q = pp + tid * PPF;
    grid_params(cx, cy, W8_, H8_, q + 0);
    grid_params(cx, cy, W16_, H16_, q + 6);
    int x0 = __float_as_int(q[0]), x1 = __float_as_int(q[1]);
    int y0 = __float_as_int(q[2]), y1 = __float_as_int(q[3]);
    float wx = q[4], wy = q[5];
    float caq[2];
#pragma unroll
    for (int c = 0; c < 2; ++c) {
      const float* cf = P.coarse_flow + ((size_t)(bi * 2 + c)) * H8_ * W8_;
      float v = cf[y0 * W8_ + x0] * (1.f - wx) * (1.f - wy)
              + cf[y0 * W8_ + x1] * wx * (1.f - wy)
              + cf[y1 * W8_ + x0] * (1.f - wx) * wy
              + cf[y1 * W8_ + x1] * wx * wy;
      caq[c] = v * 8.0f;  // scale = [W_full/W8, H_full/H8] = [8, 8]
    }
    float wpx = clipc(cx + 2.f * caq[0] / (float)WF_);
    float wpy = clipc(cy + 2.f * caq[1] / (float)HF_);
    grid_params(wpx, wpy, W8_, H8_, q + 12);
    q[18] = caq[0]; q[19] = caq[1]; q[20] = __int_as_float(bi);
    _Float16* e = ext + tid * 32;
    e[0] = (_Float16)cx; e[1] = (_Float16)cy;
    e[2] = (_Float16)(caq[0] / (float)WF_);
    e[3] = (_Float16)(caq[1] / (float)HF_);
#pragma unroll
    for (int i = 4; i < 32; ++i) e[i] = (_Float16)0.f;
  }
  __syncthreads();

  // ---------- phase 1: bilinear gathers + stage first weight ----------
  sample_feat(actB, 256, P.feat_s8,  128, H8_,  W8_,  pp, 0);   // f8
  sample_feat(actA, 256, P.ctx_s8,    64, H8_,  W8_,  pp, 0);   // fctx
  sample_feat(skp,  128, P.feat_s16, 128, H16_, W16_, pp, 6);   // f16
  sample_feat(f1w,  128, P.feat1_s8, 128, H8_,  W8_,  pp, 12);  // f1 warped
  stage_wT(wT, P.proj_ctx_w, 64, 128, 128, 64);
  prefetch_bytes(P.ffn1_w1, 128 * 256 * 4);
  __syncthreads();

  const int lane = tid & 31;
  const int g = lane >> 4;
  const int nl = lane & 15;
  const int mBase = (tid >> 5) * 16;

  // ---- L1: h1 = f8 + sig(gate1)*(fctx @ Wctx + b)  -> actB (128) ----
  for (int nt = 0; nt < 8; nt += 2) {
    v8f a0 = VZERO, a1 = VZERO;
    gemm2_acc<64, 256, 64>(a0, a1, actA, mBase, wT, 0, nt * 16);
#pragma unroll
    for (int t = 0; t < 2; ++t) {
      v8f acc = t ? a1 : a0;
      int n = (nt + t) * 16 + nl;
      float bias = P.proj_ctx_b[n];
      float gate = sigmoid_f(P.gate1[n]);
#pragma unroll
      for (int r = 0; r < 8; ++r) {
        _Float16* d = actB + (mBase + r + 8 * g) * 256 + n;
        float skipv = (float)*d;
        *d = (_Float16)(skipv + gate * (acc[r] + bias));
      }
    }
  }
  __syncthreads();

  // ---- L2a: t1 = gelu(h1 @ ffn1_w1 + b)  -> actA (256) ----
  stage_wT(wT, P.ffn1_w1, 128, 256, 256, 128);
  prefetch_bytes(P.ffn1_w2, 256 * 128 * 4);
  __syncthreads();
  for (int nt = 0; nt < 16; nt += 2) {
    v8f a0 = VZERO, a1 = VZERO;
    gemm2_acc<128, 256, 128>(a0, a1, actB, mBase, wT, 0, nt * 16);
#pragma unroll
    for (int t = 0; t < 2; ++t) {
      v8f acc = t ? a1 : a0;
      int n = (nt + t) * 16 + nl;
      float bias = P.ffn1_b1[n];
#pragma unroll
      for (int r = 0; r < 8; ++r)
        actA[(mBase + r + 8 * g) * 256 + n] = (_Float16)gelu_f(acc[r] + bias);
    }
  }
  __syncthreads();

  // ---- L2b: h2 = t1 @ ffn1_w2 + b  -> actB (128) ----
  stage_wT(wT, P.ffn1_w2, 256, 128, 128, 256);
  prefetch_bytes(P.proj_s8_w, 128 * 128 * 4);
  __syncthreads();
  for (int nt = 0; nt < 8; nt += 2) {
    v8f a0 = VZERO, a1 = VZERO;
    gemm2_acc<256, 256, 256>(a0, a1, actA, mBase, wT, 0, nt * 16);
#pragma unroll
    for (int t = 0; t < 2; ++t) {
      v8f acc = t ? a1 : a0;
      int n = (nt + t) * 16 + nl;
      float bias = P.ffn1_b2[n];
#pragma unroll
      for (int r = 0; r < 8; ++r)
        actB[(mBase + r + 8 * g) * 256 + n] = (_Float16)(acc[r] + bias);
    }
  }
  __syncthreads();

  // ---- L3: h3 = f16 + sig(gate2)*(h2 @ proj_s8 + b)  -> actA (128) ----
  stage_wT(wT, P.proj_s8_w, 128, 128, 128, 128);
  prefetch_bytes(P.ffn2_w1, 128 * 256 * 4);
  __syncthreads();
  for (int nt = 0; nt < 8; nt += 2) {
    v8f a0 = VZERO, a1 = VZERO;
    gemm2_acc<128, 256, 128>(a0, a1, actB, mBase, wT, 0, nt * 16);
#pragma unroll
    for (int t = 0; t < 2; ++t) {
      v8f acc = t ? a1 : a0;
      int n = (nt + t) * 16 + nl;
      float bias = P.proj_s8_b[n];
      float gate = sigmoid_f(P.gate2[n]);
#pragma unroll
      for (int r = 0; r < 8; ++r) {
        float skipv = (float)skp[(mBase + r + 8 * g) * 128 + n];
        actA[(mBase + r + 8 * g) * 256 + n] = (_Float16)(skipv + gate * (acc[r] + bias));
      }
    }
  }
  __syncthreads();

  // ---- L4a: t2 = gelu(h3 @ ffn2_w1 + b)  -> actB (256) ----
  stage_wT(wT, P.ffn2_w1, 128, 256, 256, 128);
  prefetch_bytes(P.ffn2_w2, 256 * 128 * 4);
  __syncthreads();
  for (int nt = 0; nt < 16; nt += 2) {
    v8f a0 = VZERO, a1 = VZERO;
    gemm2_acc<128, 256, 128>(a0, a1, actA, mBase, wT, 0, nt * 16);
#pragma unroll
    for (int t = 0; t < 2; ++t) {
      v8f acc = t ? a1 : a0;
      int n = (nt + t) * 16 + nl;
      float bias = P.ffn2_b1[n];
#pragma unroll
      for (int r = 0; r < 8; ++r)
        actB[(mBase + r + 8 * g) * 256 + n] = (_Float16)gelu_f(acc[r] + bias);
    }
  }
  __syncthreads();

  // ---- L4b: fused = t2 @ ffn2_w2 + b  -> actA (128) ----
  stage_wT(wT, P.ffn2_w2, 256, 128, 128, 256);
  prefetch_bytes(P.hw0, 260 * 256 * 4);
  __syncthreads();
  for (int nt = 0; nt < 8; nt += 2) {
    v8f a0 = VZERO, a1 = VZERO;
    gemm2_acc<256, 256, 256>(a0, a1, actB, mBase, wT, 0, nt * 16);
#pragma unroll
    for (int t = 0; t < 2; ++t) {
      v8f acc = t ? a1 : a0;
      int n = (nt + t) * 16 + nl;
      float bias = P.ffn2_b2[n];
#pragma unroll
      for (int r = 0; r < 8; ++r)
        actA[(mBase + r + 8 * g) * 256 + n] = (_Float16)(acc[r] + bias);
    }
  }
  __syncthreads();

  // ---- H0: h = relu([fused | f1w | extras] @ hw0 + b)  -> actB (256) ----
  stage_wT(wT, P.hw0, 260, 256, 256, 288);   // rows 260..287 zero-padded
  prefetch_bytes(P.hw1, 256 * 128 * 4);
  __syncthreads();
  for (int nt = 0; nt < 16; nt += 2) {
    v8f a0 = VZERO, a1 = VZERO;
    gemm2_acc<128, 256, 288>(a0, a1, actA, mBase, wT, 0,   nt * 16);
    gemm2_acc<128, 128, 288>(a0, a1, f1w,  mBase, wT, 128, nt * 16);
    gemm2_acc<32,  32,  288>(a0, a1, ext,  mBase, wT, 256, nt * 16);
#pragma unroll
    for (int t = 0; t < 2; ++t) {
      v8f acc = t ? a1 : a0;
      int n = (nt + t) * 16 + nl;
      float bias = P.hb0[n];
#pragma unroll
      for (int r = 0; r < 8; ++r)
        actB[(mBase + r + 8 * g) * 256 + n] = (_Float16)fmaxf(acc[r] + bias, 0.f);
    }
  }
  __syncthreads();

  // ---- H1: h = relu(h @ hw1 + b)  -> actA (128) ----
  stage_wT(wT, P.hw1, 256, 128, 128, 256);
  prefetch_bytes(P.hw2, 128 * 64 * 4);
  __syncthreads();
  for (int nt = 0; nt < 8; nt += 2) {
    v8f a0 = VZERO, a1 = VZERO;
    gemm2_acc<256, 256, 256>(a0, a1, actB, mBase, wT, 0, nt * 16);
#pragma unroll
    for (int t = 0; t < 2; ++t) {
      v8f acc = t ? a1 : a0;
      int n = (nt + t) * 16 + nl;
      float bias = P.hb1[n];
#pragma unroll
      for (int r = 0; r < 8; ++r)
        actA[(mBase + r + 8 * g) * 256 + n] = (_Float16)fmaxf(acc[r] + bias, 0.f);
    }
  }
  __syncthreads();

  // ---- H2: h = relu(h @ hw2 + b)  -> actB (64) ----
  stage_wT(wT, P.hw2, 128, 64, 64, 128);
  prefetch_bytes(P.hw3, 64 * 2 * 4);
  __syncthreads();
  for (int nt = 0; nt < 4; nt += 2) {
    v8f a0 = VZERO, a1 = VZERO;
    gemm2_acc<128, 256, 128>(a0, a1, actA, mBase, wT, 0, nt * 16);
#pragma unroll
    for (int t = 0; t < 2; ++t) {
      v8f acc = t ? a1 : a0;
      int n = (nt + t) * 16 + nl;
      float bias = P.hb2[n];
#pragma unroll
      for (int r = 0; r < 8; ++r)
        actB[(mBase + r + 8 * g) * 256 + n] = (_Float16)fmaxf(acc[r] + bias, 0.f);
    }
  }
  __syncthreads();

  // ---- H3: delta = h @ hw3 + b ; flow = coarse + delta*[W,H] -> global out ----
  stage_wT(wT, P.hw3, 64, 2, 16, 64);        // N padded 2 -> 16 with zeros
  __syncthreads();
  {
    v8f acc = VZERO;
    gemm_acc<64, 256, 64>(acc, actB, mBase, wT, 0, 0);
    if (nl < 2) {
      float bias = P.hb3[nl];
      float sc = (nl == 0) ? (float)WF_ : (float)HF_;
#pragma unroll
      for (int r = 0; r < 8; ++r) {
        int m = mBase + r + 8 * g;
        int p = tileBase + m;
        int bi = p / (HF_ * WF_);
        int rem = p - bi * (HF_ * WF_);
        int h = rem >> 9;
        int w = rem & (WF_ - 1);
        float caq = pp[m * PPF + 18 + nl];
        float flow = caq + (acc[r] + bias) * sc;
        P.out[(((size_t)bi * 2 + nl) * HF_ + h) * WF_ + w] = flow;
      }
    }
  }
}

extern "C" void kernel_launch(void* const* d_in, const int* in_sizes, int n_in,
                              void* d_out, int out_size, void* d_ws, size_t ws_size,
                              hipStream_t stream) {
  (void)in_sizes; (void)n_in; (void)out_size; (void)d_ws; (void)ws_size;
  Params P;
  P.feat_s8     = (const float*)d_in[1];
  P.feat1_s8    = (const float*)d_in[2];
  P.feat_s16    = (const float*)d_in[3];
  P.ctx_s8      = (const float*)d_in[4];
  P.coarse_flow = (const float*)d_in[5];
  P.proj_ctx_w  = (const float*)d_in[6];  P.proj_ctx_b = (const float*)d_in[7];
  P.gate1       = (const float*)d_in[8];
  P.ffn1_w1     = (const float*)d_in[9];  P.ffn1_b1 = (const float*)d_in[10];
  P.ffn1_w2     = (const float*)d_in[11]; P.ffn1_b2 = (const float*)d_in[12];
  P.proj_s8_w   = (const float*)d_in[13]; P.proj_s8_b = (const float*)d_in[14];
  P.gate2       = (const float*)d_in[15];
  P.ffn2_w1     = (const float*)d_in[16]; P.ffn2_b1 = (const float*)d_in[17];
  P.ffn2_w2     = (const float*)d_in[18]; P.ffn2_b2 = (const float*)d_in[19];
  P.hw0 = (const float*)d_in[20]; P.hb0 = (const float*)d_in[21];
  P.hw1 = (const float*)d_in[22]; P.hb1 = (const float*)d_in[23];
  P.hw2 = (const float*)d_in[24]; P.hb2 = (const float*)d_in[25];
  P.hw3 = (const float*)d_in[26]; P.hb3 = (const float*)d_in[27];
  P.out = (float*)d_out;

  dim3 grid(NPIX / TILE);
  dim3 block(NTHR);
  hipLaunchKernelGGL(ifd_kernel, grid, block, LDS_BYTES, stream, P);
}
